// GQAttention_88751204205080
// MI455X (gfx1250) — matrix operand order repaired and checked by reading, not compiled, not run on hardware
//
#include <hip/hip_runtime.h>
#include <stdint.h>

#define NH   32
#define NKV  8
#define HD   64
#define DIM  2048
#define BATCH 4
#define SEQ  1024

typedef __attribute__((ext_vector_type(16))) __bf16 v16bf;
typedef __attribute__((ext_vector_type(8)))  float  v8f;

union Frag32 { uint4 u[2]; v16bf v; };
union Chunk8 { uint4 q; unsigned short s[8]; };

__device__ __forceinline__ unsigned short f2bf(float f) {
    unsigned int u = __float_as_uint(f);
    unsigned int r = u + 0x7FFFu + ((u >> 16) & 1u);   // round-to-nearest-even
    return (unsigned short)(r >> 16);
}
__device__ __forceinline__ float bf2f(unsigned short h) {
    return __uint_as_float(((unsigned int)h) << 16);
}

// ---------------------------------------------------------------- cast f32->bf16
__global__ void cast_f32_to_bf16(const float* __restrict__ in,
                                 unsigned short* __restrict__ out, int n) {
    int i = blockIdx.x * blockDim.x + threadIdx.x;
    if (i < n) out[i] = f2bf(in[i]);
}

// ---------------------------------------------------------------- WMMA GEMM
// C[M,N] = A[M,K](bf16, row-major) * W[N,K](bf16, row-major)^T
// One wave computes a 32(M) x 64(N) tile: B fragments reused across 2 M
// sub-tiles, A fragments reused across 4 N sub-tiles (8 WMMA / k-step).
template<bool OUT_BF16>
__global__ __launch_bounds__(256)
void gemm_bf16_wmma(const unsigned short* __restrict__ A,
                    const unsigned short* __restrict__ W,
                    void* __restrict__ Cv, int M, int N, int K) {
    int wave   = (blockIdx.x * 256 + threadIdx.x) >> 5;
    int lane   = threadIdx.x & 31;
    int tilesN = N >> 6;
    int total  = (M >> 5) * tilesN;
    if (wave >= total) return;
    int tm = wave / tilesN, tn = wave % tilesN;
    int n  = lane & 15, g = lane >> 4;

    v8f acc[2][4];
#pragma unroll
    for (int i = 0; i < 2; ++i)
#pragma unroll
        for (int t = 0; t < 4; ++t) acc[i][t] = (v8f){0.f,0.f,0.f,0.f,0.f,0.f,0.f,0.f};

    const unsigned short* arow0 = A + (size_t)(tm * 32 + n) * K;        // A rows m, m+16
    const unsigned short* arow1 = arow0 + (size_t)16 * K;
    const unsigned short* wrow  = W + (size_t)(tn * 64 + n) * K;        // B col n = lane&15

    for (int kk = 0; kk < K; kk += 32) {
        if (kk + 256 < K) {                       // stream-ahead into L2/WGP$
            __builtin_prefetch(arow0 + kk + 256, 0, 3);
            __builtin_prefetch(wrow  + kk + 256, 0, 3);
        }
        Frag32 a0, a1;
        a0.u[0] = *(const uint4*)(arow0 + kk + 16 * g);
        a0.u[1] = *(const uint4*)(arow0 + kk + 16 * g + 8);
        a1.u[0] = *(const uint4*)(arow1 + kk + 16 * g);
        a1.u[1] = *(const uint4*)(arow1 + kk + 16 * g + 8);
        const unsigned short* wp = wrow + kk + 16 * g;
        v16bf b0 = *(const v16bf*)(wp);
        v16bf b1 = *(const v16bf*)(wp + (size_t)16 * K);
        v16bf b2 = *(const v16bf*)(wp + (size_t)32 * K);
        v16bf b3 = *(const v16bf*)(wp + (size_t)48 * K);
        acc[0][0] = __builtin_amdgcn_wmma_f32_16x16x32_bf16(false, a0.v, false, b0, (short)0, acc[0][0], false, false);
        acc[0][1] = __builtin_amdgcn_wmma_f32_16x16x32_bf16(false, a0.v, false, b1, (short)0, acc[0][1], false, false);
        acc[0][2] = __builtin_amdgcn_wmma_f32_16x16x32_bf16(false, a0.v, false, b2, (short)0, acc[0][2], false, false);
        acc[0][3] = __builtin_amdgcn_wmma_f32_16x16x32_bf16(false, a0.v, false, b3, (short)0, acc[0][3], false, false);
        acc[1][0] = __builtin_amdgcn_wmma_f32_16x16x32_bf16(false, a1.v, false, b0, (short)0, acc[1][0], false, false);
        acc[1][1] = __builtin_amdgcn_wmma_f32_16x16x32_bf16(false, a1.v, false, b1, (short)0, acc[1][1], false, false);
        acc[1][2] = __builtin_amdgcn_wmma_f32_16x16x32_bf16(false, a1.v, false, b2, (short)0, acc[1][2], false, false);
        acc[1][3] = __builtin_amdgcn_wmma_f32_16x16x32_bf16(false, a1.v, false, b3, (short)0, acc[1][3], false, false);
    }

    int col0 = tn * 64 + n;
#pragma unroll
    for (int i = 0; i < 2; ++i) {
#pragma unroll
        for (int r = 0; r < 8; ++r) {
            int row = tm * 32 + i * 16 + r + 8 * g;
            size_t base = (size_t)row * N + col0;
            if (OUT_BF16) {
                unsigned short* C = (unsigned short*)Cv;
                C[base]      = f2bf(acc[i][0][r]);
                C[base + 16] = f2bf(acc[i][1][r]);
                C[base + 32] = f2bf(acc[i][2][r]);
                C[base + 48] = f2bf(acc[i][3][r]);
            } else {
                float* C = (float*)Cv;
                C[base]      = acc[i][0][r];
                C[base + 16] = acc[i][1][r];
                C[base + 32] = acc[i][2][r];
                C[base + 48] = acc[i][3][r];
            }
        }
    }
}

// ---------------------------------------------------------------- RoPE + permute
// in: [B*T, nh*64] token-major (GEMM output)  ->  out: [B, nh, T, 64]
__global__ void rope_perm(const unsigned short* __restrict__ in,
                          const float* __restrict__ cosb,
                          const float* __restrict__ sinb,
                          unsigned short* __restrict__ out, int nh) {
    int i = blockIdx.x * blockDim.x + threadIdx.x;
    int total = BATCH * SEQ * nh * 32;
    if (i >= total) return;
    int p = i & 31; int r = i >> 5;
    int h = r % nh; r /= nh;
    int t = r % SEQ; int b = r / SEQ;
    size_t src = (size_t)(b * SEQ + t) * (nh * 64) + h * 64 + 2 * p;
    float xe = bf2f(in[src]);
    float xo = bf2f(in[src + 1]);
    float c = cosb[t * 32 + p], s = sinb[t * 32 + p];
    size_t dst = ((size_t)(b * nh + h) * SEQ + t) * 64 + 2 * p;
    out[dst]     = f2bf(xe * c - xo * s);
    out[dst + 1] = f2bf(xe * s + xo * c);
}

// in: [B*T, NKV*64] -> out: [B, NKV, T, 64]
__global__ void v_perm(const unsigned short* __restrict__ in,
                       unsigned short* __restrict__ out) {
    int i = blockIdx.x * blockDim.x + threadIdx.x;
    int total = BATCH * SEQ * NKV * 64;
    if (i >= total) return;
    int d = i & 63; int r = i >> 6;
    int h = r % NKV; r /= NKV;
    int t = r % SEQ; int b = r / SEQ;
    out[((size_t)(b * NKV + h) * SEQ + t) * 64 + d] =
        in[(size_t)(b * SEQ + t) * (NKV * 64) + h * 64 + d];
}

// ---------------------------------------------------------------- Flash attention
// grid: B*NH*(SEQ/64) blocks of 128 threads (4 waves, 16 query rows each)
__global__ __launch_bounds__(128)
void flash_attn_wmma(const unsigned short* __restrict__ Q,   // [B,NH,SEQ,64]
                     const unsigned short* __restrict__ Kh,  // [B,NKV,SEQ,64]
                     const unsigned short* __restrict__ Vh,  // [B,NKV,SEQ,64]
                     unsigned short* __restrict__ O) {       // [B*SEQ, NH*64]
    __shared__ unsigned short sK[64 * 64];      // [key][d]   (== B layout for QK^T)
    __shared__ unsigned short sV[64 * 64];      // [d][key]   (transposed for PV)
    __shared__ unsigned short sP[4 * 16 * 64];  // wave-private P tiles

    int bid = blockIdx.x;
    int qt = bid & 15; bid >>= 4;       // SEQ/64 = 16 query tiles
    int h  = bid & 31; int b = bid >> 5;
    int kvh = h >> 2;                   // GQA: 32 heads -> 8 kv heads
    int tid = threadIdx.x;
    int wid = tid >> 5, lane = tid & 31;
    int n = lane & 15, g = lane >> 4;

    const unsigned short* qbase = Q  + (size_t)(b * NH  + h)   * SEQ * 64;
    const unsigned short* kbase = Kh + (size_t)(b * NKV + kvh) * SEQ * 64;
    const unsigned short* vbase = Vh + (size_t)(b * NKV + kvh) * SEQ * 64;

    // raw LDS byte address of sK (flat shared addr low 32 bits == LDS offset)
    unsigned sKlds = (unsigned)(uintptr_t)(&sK[0]);

    int qrow0 = qt * 64 + wid * 16;

    // Q A-fragments for d = [0,32) and [32,64)
    Frag32 qa0, qa1;
    {
        const unsigned short* qp = qbase + (size_t)(qrow0 + n) * 64;
        qa0.u[0] = *(const uint4*)(qp + 16 * g);
        qa0.u[1] = *(const uint4*)(qp + 16 * g + 8);
        qa1.u[0] = *(const uint4*)(qp + 32 + 16 * g);
        qa1.u[1] = *(const uint4*)(qp + 32 + 16 * g + 8);
    }

    float mrow[8], lrow[8];
#pragma unroll
    for (int r = 0; r < 8; ++r) { mrow[r] = -__builtin_inff(); lrow[r] = 0.f; }
    v8f o0 = {}, o1 = {}, o2 = {}, o3 = {};

    for (int kb = 0; kb <= qt; ++kb) {
        __syncthreads();
        // stage K tile: direct global -> LDS via CDNA5 async DMA path
        for (int i = tid; i < 512; i += 128) {
            unsigned lds = sKlds + (unsigned)i * 16u;
            unsigned long long ga =
                (unsigned long long)(uintptr_t)(kbase + (size_t)kb * 4096 + (size_t)i * 8);
            asm volatile("global_load_async_to_lds_b128 %0, %1, off"
                         :: "v"(lds), "v"(ga) : "memory");
        }
        // stage V tile transposed [d][key] (vectorized read, scatter store)
        for (int i = tid; i < 512; i += 128) {
            int key = i >> 3;
            int dc  = (i & 7) * 8;
            Chunk8 c8;
            c8.q = *(const uint4*)(vbase + (size_t)(kb * 64 + key) * 64 + dc);
#pragma unroll
            for (int j = 0; j < 8; ++j) sV[(dc + j) * 64 + key] = c8.s[j];
        }
        asm volatile("s_wait_asynccnt 0x0" ::: "memory");
        __syncthreads();

        // S = Q * K^T  (16 x 64 keys per wave)
        v8f s0 = {}, s1 = {}, s2 = {}, s3 = {};
#pragma unroll
        for (int kk = 0; kk < 2; ++kk) {
            v16bf a = kk ? qa1.v : qa0.v;
            const unsigned short* kp = &sK[kk * 32 + 16 * g];
            v16bf b0 = *(const v16bf*)(kp + (0 * 16 + n) * 64);
            v16bf b1 = *(const v16bf*)(kp + (1 * 16 + n) * 64);
            v16bf b2 = *(const v16bf*)(kp + (2 * 16 + n) * 64);
            v16bf b3 = *(const v16bf*)(kp + (3 * 16 + n) * 64);
            s0 = __builtin_amdgcn_wmma_f32_16x16x32_bf16(false, a, false, b0, (short)0, s0, false, false);
            s1 = __builtin_amdgcn_wmma_f32_16x16x32_bf16(false, a, false, b1, (short)0, s1, false, false);
            s2 = __builtin_amdgcn_wmma_f32_16x16x32_bf16(false, a, false, b2, (short)0, s2, false, false);
            s3 = __builtin_amdgcn_wmma_f32_16x16x32_bf16(false, a, false, b3, (short)0, s3, false, false);
        }

        // scale + causal mask
        float sv[4][8];
#pragma unroll
        for (int t = 0; t < 4; ++t) {
            int keyg = kb * 64 + t * 16 + n;
#pragma unroll
            for (int r = 0; r < 8; ++r) {
                int qr = qrow0 + r + 8 * g;
                float x = (t == 0 ? s0[r] : t == 1 ? s1[r] : t == 2 ? s2[r] : s3[r]) * 0.125f;
                sv[t][r] = (keyg > qr) ? -__builtin_inff() : x;
            }
        }

        // online softmax (row reductions across the 16 lanes holding a row)
#pragma unroll
        for (int r = 0; r < 8; ++r) {
            float mnew = fmaxf(fmaxf(sv[0][r], sv[1][r]), fmaxf(sv[2][r], sv[3][r]));
#pragma unroll
            for (int off = 1; off < 16; off <<= 1)
                mnew = fmaxf(mnew, __shfl_xor(mnew, off, 32));
            mnew = fmaxf(mnew, mrow[r]);
            float corr = __expf(mrow[r] - mnew);
            float rs = 0.f;
#pragma unroll
            for (int t = 0; t < 4; ++t) { float p = __expf(sv[t][r] - mnew); sv[t][r] = p; rs += p; }
#pragma unroll
            for (int off = 1; off < 16; off <<= 1)
                rs += __shfl_xor(rs, off, 32);
            lrow[r] = lrow[r] * corr + rs;
            mrow[r] = mnew;
            o0[r] *= corr; o1[r] *= corr; o2[r] *= corr; o3[r] *= corr;
        }

        // C-layout -> A-layout round trip through wave-private LDS
        unsigned short* pp = &sP[wid * 1024];
#pragma unroll
        for (int t = 0; t < 4; ++t)
#pragma unroll
            for (int r = 0; r < 8; ++r)
                pp[(r + 8 * g) * 64 + t * 16 + n] = f2bf(sv[t][r]);
        asm volatile("s_wait_dscnt 0x0" ::: "memory");   // CDNA5 split DS counter

        // O += P * V
#pragma unroll
        for (int kk = 0; kk < 2; ++kk) {
            Frag32 pa;
            const unsigned short* pr = pp + n * 64 + kk * 32 + 16 * g;  // A row m = lane&15
            pa.u[0] = *(const uint4*)(pr);
            pa.u[1] = *(const uint4*)(pr + 8);
            const unsigned short* vp = &sV[kk * 32 + 16 * g];
            v16bf b0 = *(const v16bf*)(vp + (0 * 16 + n) * 64);
            v16bf b1 = *(const v16bf*)(vp + (1 * 16 + n) * 64);
            v16bf b2 = *(const v16bf*)(vp + (2 * 16 + n) * 64);
            v16bf b3 = *(const v16bf*)(vp + (3 * 16 + n) * 64);
            o0 = __builtin_amdgcn_wmma_f32_16x16x32_bf16(false, pa.v, false, b0, (short)0, o0, false, false);
            o1 = __builtin_amdgcn_wmma_f32_16x16x32_bf16(false, pa.v, false, b1, (short)0, o1, false, false);
            o2 = __builtin_amdgcn_wmma_f32_16x16x32_bf16(false, pa.v, false, b2, (short)0, o2, false, false);
            o3 = __builtin_amdgcn_wmma_f32_16x16x32_bf16(false, pa.v, false, b3, (short)0, o3, false, false);
        }
    }

    // write attention output token-major [B*T, NH*64] as bf16 (feeds Wo GEMM)
#pragma unroll
    for (int r = 0; r < 8; ++r) {
        int qr = qrow0 + r + 8 * g;
        float inv = 1.f / lrow[r];
        size_t base = (size_t)(b * SEQ + qr) * (NH * 64) + h * 64 + n;
        O[base]      = f2bf(o0[r] * inv);
        O[base + 16] = f2bf(o1[r] * inv);
        O[base + 32] = f2bf(o2[r] * inv);
        O[base + 48] = f2bf(o3[r] * inv);
    }
}

// ---------------------------------------------------------------- launch
extern "C" void kernel_launch(void* const* d_in, const int* in_sizes, int n_in,
                              void* d_out, int out_size, void* d_ws, size_t ws_size,
                              hipStream_t stream) {
    (void)in_sizes; (void)n_in; (void)out_size; (void)ws_size;
    const float* x    = (const float*)d_in[0];
    const float* cosb = (const float*)d_in[1];
    const float* sinb = (const float*)d_in[2];
    const float* Wq   = (const float*)d_in[3];
    const float* Wk   = (const float*)d_in[4];
    const float* Wv   = (const float*)d_in[5];
    const float* Wo   = (const float*)d_in[6];
    float* out = (float*)d_out;

    const int M = BATCH * SEQ;                 // 4096 tokens
    char* ws = (char*)d_ws;
    size_t off = 0;
    auto alloc = [&](size_t bytes) -> void* {
        void* p = ws + off;
        off += (bytes + 255) & ~(size_t)255;
        return p;
    };
    unsigned short* xb   = (unsigned short*)alloc((size_t)M * DIM * 2);
    unsigned short* wqb  = (unsigned short*)alloc((size_t)NH  * HD * DIM * 2);
    unsigned short* wkb  = (unsigned short*)alloc((size_t)NKV * HD * DIM * 2);
    unsigned short* wvb  = (unsigned short*)alloc((size_t)NKV * HD * DIM * 2);
    unsigned short* wob  = (unsigned short*)alloc((size_t)DIM * NH * HD * 2);
    unsigned short* qtmp = (unsigned short*)alloc((size_t)M * NH  * HD * 2);
    unsigned short* ktmp = (unsigned short*)alloc((size_t)M * NKV * HD * 2);
    unsigned short* vtmp = (unsigned short*)alloc((size_t)M * NKV * HD * 2);
    unsigned short* qbuf = (unsigned short*)alloc((size_t)M * NH  * HD * 2);
    unsigned short* kbuf = (unsigned short*)alloc((size_t)M * NKV * HD * 2);
    unsigned short* vbuf = (unsigned short*)alloc((size_t)M * NKV * HD * 2);
    unsigned short* obuf = (unsigned short*)alloc((size_t)M * NH  * HD * 2);

    auto cast = [&](const float* src, unsigned short* dst, int nelem) {
        cast_f32_to_bf16<<<(nelem + 255) / 256, 256, 0, stream>>>(src, dst, nelem);
    };
    cast(x,  xb,  M * DIM);
    cast(Wq, wqb, NH  * HD * DIM);
    cast(Wk, wkb, NKV * HD * DIM);
    cast(Wv, wvb, NKV * HD * DIM);
    cast(Wo, wob, DIM * NH * HD);

    auto gemm_blocks = [](int m, int nn) { return ((m / 32) * (nn / 64) + 7) / 8; };

    // QKV projections (bf16 out)
    gemm_bf16_wmma<true><<<gemm_blocks(M, NH * HD), 256, 0, stream>>>(xb, wqb, qtmp, M, NH * HD, DIM);
    gemm_bf16_wmma<true><<<gemm_blocks(M, NKV * HD), 256, 0, stream>>>(xb, wkb, ktmp, M, NKV * HD, DIM);
    gemm_bf16_wmma<true><<<gemm_blocks(M, NKV * HD), 256, 0, stream>>>(xb, wvb, vtmp, M, NKV * HD, DIM);

    // RoPE + head-major permutes
    {
        int nq = BATCH * SEQ * NH * 32;
        rope_perm<<<(nq + 255) / 256, 256, 0, stream>>>(qtmp, cosb, sinb, qbuf, NH);
        int nk = BATCH * SEQ * NKV * 32;
        rope_perm<<<(nk + 255) / 256, 256, 0, stream>>>(ktmp, cosb, sinb, kbuf, NKV);
        int nv = BATCH * SEQ * NKV * 64;
        v_perm<<<(nv + 255) / 256, 256, 0, stream>>>(vtmp, vbuf);
    }

    // flash attention
    flash_attn_wmma<<<BATCH * NH * (SEQ / 64), 128, 0, stream>>>(qbuf, kbuf, vbuf, obuf);

    // output projection (f32 out)
    gemm_bf16_wmma<false><<<gemm_blocks(M, DIM), 256, 0, stream>>>(obuf, wob, out, M, DIM, DIM);
}
